// _BambaTextModel_22797686407732
// MI455X (gfx1250) — matrix operand order repaired and checked by reading, not compiled, not run on hardware
//
#include <hip/hip_runtime.h>
#include <hip/hip_bf16.h>

// ---------------------------------------------------------------------------
// Types for CDNA5 WMMA
// ---------------------------------------------------------------------------
typedef __attribute__((ext_vector_type(16))) __bf16 v16bf;
typedef __attribute__((ext_vector_type(8)))  float  v8f;

// Native f32 -> bf16 conversion (RNE): clang lowers fptrunc to hardware cvt.
__device__ __forceinline__ unsigned int bf16_pack2(float lo, float hi) {
  union { __bf16 h[2]; unsigned int u; } r;
  r.h[0] = (__bf16)lo;
  r.h[1] = (__bf16)hi;
  return r.u;
}

__device__ __forceinline__ float silu_f(float x) {
  return x / (1.0f + __expf(-x));
}

// ---------------------------------------------------------------------------
// Model constants
// ---------------------------------------------------------------------------
#define SEQ     1024
#define DMODEL  1024
#define DFF_    4096
#define DIN     2048          // D_INNER
#define GN      128           // N_GROUPS * D_STATE
#define NPROJ   4384          // 2*DIN + 2*GN + 32
#define CONVD   2304          // DIN + 2*GN
#define MHEADS  32
#define MDH     64
#define DSTATE  128
#define AH      16            // attention heads
#define AHKV    4
#define ADH     64

// ---------------------------------------------------------------------------
// GEMM: C[M,N] = A[M,K] (f32) @ B[K,N] (f32), bf16 WMMA + f32 accumulate.
// NN/KK are compile-time so all B strides become immediate IOFFSETs
// (<= 15*NN*4 bytes, well within the signed 24-bit field) and the k-loop is
// fully strength-reduced. M is always SEQ (1024). The ragged N edge
// (NN % 128 != 0, only in_proj) uses EDGE=true with per-thread-uniform
// column guards only.
//
// Tiles are staged into LDS PRE-PERMUTED into a wave32 fragment
// layout with k = 16*laneHalf + s per lane, so each fragment read is two ds_load_b128.
// Macro tile 64x128, BK=32, 256 threads (8 waves), wave tile 32x32.
// ---------------------------------------------------------------------------
#define BM 64
#define BN 128
#define BK 32

template <int NN, int KK, bool EDGE>
__launch_bounds__(256)
__global__ void gemm_wmma_bf16(const float* __restrict__ A,
                               const float* __restrict__ B,
                               float* __restrict__ C, int n0) {
  // packed fragments: slot = lane, 16 bf16 (32B) contiguous per lane
  __shared__ uint4 pA[4 * 32 * 2];   // 4 m-subtiles x 32 lanes x 32B
  __shared__ uint4 pB[8 * 32 * 2];   // 8 n-subtiles x 32 lanes x 32B

  const int tid      = threadIdx.x;
  const int wave     = tid >> 5;
  const int lane     = tid & 31;
  const int lane15   = lane & 15;
  const int laneHalf = lane >> 4;

  const int blockM   = blockIdx.y * BM;
  const int blockN   = n0 + blockIdx.x * BN;
  const int waveMoff = (wave >> 2) * 32;   // 2 wave rows
  const int waveNoff = (wave & 3)  * 32;   // 4 wave cols
  const int mSub0    = waveMoff >> 4;
  const int nSub0    = waveNoff >> 4;

  // --- A staging assignment: thread -> 8 consecutive K of one row ---
  // A 16x32 layout: lanes 0-15 m=lane: u[0..15]=K0..15
  //                 lanes16-31 m=lane-16: u[0..15]=K16..31
  const int aRow  = tid >> 2;
  const int aKb   = (tid & 3) * 8;
  const int aSub  = aRow >> 4;
  const int aSlot = (aRow & 15) + ((aKb & 16) ? 16 : 0);
  const int aHalf = (aKb & 8) ? 1 : 0;
  uint4* const aDst = &pA[(aSub * 32 + aSlot) * 2 + aHalf];
  const float* const aSrcRow = A + (size_t)(blockM + aRow) * KK + aKb;

  // --- B staging assignment: thread -> one (n,KHalf) column strip of 16 ---
  // B 32x16 layout used here (contiguous k): lane15 = N col; lanes 0-15 K0..15, lanes 16-31 K16..31
  const int bSub  = tid >> 5;         // 0..7  (n subtile)
  const int bSlot = tid & 31;         // dest lane slot
  const int bHalf = bSlot >> 4;       // K half
  const int bN    = bSub * 16 + (bSlot & 15);
  uint4* const bDst = &pB[(bSub * 32 + bSlot) * 2];
  const bool bValid = !EDGE || (blockN + bN < NN);
  const float* const bSrcCol = B + (size_t)bHalf * 16 * NN + blockN + bN;

  v8f acc[2][2];
  #pragma unroll
  for (int i = 0; i < 2; ++i)
    #pragma unroll
    for (int j = 0; j < 2; ++j)
      acc[i][j] = (v8f){0.f,0.f,0.f,0.f,0.f,0.f,0.f,0.f};

  for (int k0 = 0; k0 < KK; k0 += BK) {
    // ---- stage A: 2 x float4 load, pack, 1 x ds_store_b128 ----
    {
      const float4 f0 = *(const float4*)(aSrcRow + k0);
      const float4 f1 = *(const float4*)(aSrcRow + k0 + 4);
      uint4 r;
      r.x = bf16_pack2(f0.x, f0.y);
      r.y = bf16_pack2(f0.z, f0.w);
      r.z = bf16_pack2(f1.x, f1.y);
      r.w = bf16_pack2(f1.z, f1.w);
      *aDst = r;
    }
    // ---- stage B: 16 K-strided dwords, constant offsets e*NN (imm) ----
    {
      float f[16];
      if (bValid) {
        const float* s = bSrcCol + (size_t)k0 * NN;
        #pragma unroll
        for (int e = 0; e < 16; ++e) f[e] = s[e * NN];
      } else {
        #pragma unroll
        for (int e = 0; e < 16; ++e) f[e] = 0.0f;
      }
      uint4 r0, r1;
      r0.x = bf16_pack2(f[0],  f[1]);  r0.y = bf16_pack2(f[2],  f[3]);
      r0.z = bf16_pack2(f[4],  f[5]);  r0.w = bf16_pack2(f[6],  f[7]);
      r1.x = bf16_pack2(f[8],  f[9]);  r1.y = bf16_pack2(f[10], f[11]);
      r1.z = bf16_pack2(f[12], f[13]); r1.w = bf16_pack2(f[14], f[15]);
      bDst[0] = r0;
      bDst[1] = r1;
    }
    // prefetch next K-slab while this one is consumed
    if (k0 + BK < KK) {
      __builtin_prefetch(aSrcRow + k0 + BK, 0, 1);
      if (bValid) __builtin_prefetch(bSrcCol + (size_t)(k0 + BK) * NN, 0, 1);
    }
    __syncthreads();

    // ---- fragments: 2 x ds_load_b128 each ----
    union FV { uint4 q[2]; v16bf v; };
    FV a[2], b[2];
    #pragma unroll
    for (int t = 0; t < 2; ++t) {
      a[t].q[0] = pA[((mSub0 + t) * 32 + lane) * 2];
      a[t].q[1] = pA[((mSub0 + t) * 32 + lane) * 2 + 1];
      b[t].q[0] = pB[((nSub0 + t) * 32 + lane) * 2];
      b[t].q[1] = pB[((nSub0 + t) * 32 + lane) * 2 + 1];
    }

    #pragma unroll
    for (int i = 0; i < 2; ++i)
      #pragma unroll
      for (int j = 0; j < 2; ++j)
        acc[i][j] = __builtin_amdgcn_wmma_f32_16x16x32_bf16(
            false, a[i].v, false, b[j].v, (short)0, acc[i][j], false, false);

    __syncthreads();
  }

  // ---- store (C/D layout: lane15 = N, VGPR r -> M = laneHalf*8 + r) ----
  #pragma unroll
  for (int i = 0; i < 2; ++i) {
    #pragma unroll
    for (int j = 0; j < 2; ++j) {
      const int col = blockN + waveNoff + j * 16 + lane15;
      if (EDGE && col >= NN) continue;         // per-thread uniform guard
      const int rowBase = blockM + waveMoff + i * 16 + laneHalf * 8;
      #pragma unroll
      for (int r = 0; r < 8; ++r)
        C[(size_t)(rowBase + r) * NN + col] = acc[i][j][r];
    }
  }
}

template <int NN, int KK>
static inline void run_gemm(const float* A, const float* B, float* C,
                            hipStream_t s) {
  constexpr int nFullTiles = NN / BN;
  constexpr int nFull = nFullTiles * BN;
  if (nFullTiles > 0) {
    dim3 grid(nFullTiles, SEQ / BM);
    gemm_wmma_bf16<NN, KK, false><<<grid, 256, 0, s>>>(A, B, C, 0);
  }
  if (NN - nFull > 0) {
    dim3 grid(1, SEQ / BM);
    gemm_wmma_bf16<NN, KK, true><<<grid, 256, 0, s>>>(A, B, C, nFull);
  }
}

// ---------------------------------------------------------------------------
// Embedding gather
// ---------------------------------------------------------------------------
__global__ void embed_kernel(const float* __restrict__ embed,
                             const int* __restrict__ ids,
                             float* __restrict__ h) {
  const int s = blockIdx.x;
  const int row = ids[s];
  for (int d = threadIdx.x; d < DMODEL; d += blockDim.x)
    h[(size_t)s * DMODEL + d] = embed[(size_t)row * DMODEL + d];
}

// ---------------------------------------------------------------------------
// RMSNorm (one block per row)
// ---------------------------------------------------------------------------
__global__ void rmsnorm_kernel(const float* __restrict__ x,
                               const float* __restrict__ w,
                               float* __restrict__ out, int D) {
  const int row = blockIdx.x;
  const float* xr = x + (size_t)row * D;
  float ss = 0.f;
  for (int d = threadIdx.x; d < D; d += blockDim.x) { float v = xr[d]; ss += v * v; }
  __shared__ float red[256];
  red[threadIdx.x] = ss; __syncthreads();
  for (int st = 128; st > 0; st >>= 1) {
    if ((int)threadIdx.x < st) red[threadIdx.x] += red[threadIdx.x + st];
    __syncthreads();
  }
  const float inv = rsqrtf(red[0] / (float)D + 1e-6f);
  for (int d = threadIdx.x; d < D; d += blockDim.x)
    out[(size_t)row * D + d] = xr[d] * inv * w[d];
}

// ---------------------------------------------------------------------------
// RoPE in place on [S, nH, 64]; pair (i, i+32) shares frequency i
// ---------------------------------------------------------------------------
__global__ void rope_kernel(float* __restrict__ x, const int* __restrict__ pos,
                            int nH) {
  const int idx = blockIdx.x * blockDim.x + threadIdx.x;
  const int total = SEQ * nH * 32;
  if (idx >= total) return;
  const int i  = idx & 31;
  const int hh = (idx >> 5) % nH;
  const int s  = idx / (32 * nH);
  const float p = (float)pos[s];
  const float freq = __powf(10000.0f, -(float)(2 * i) / 64.0f);
  const float ang = p * freq;
  const float c = __cosf(ang), sn = __sinf(ang);
  const size_t base = ((size_t)s * nH + hh) * 64;
  const float x1 = x[base + i];
  const float x2 = x[base + 32 + i];
  x[base + i]      = x1 * c - x2 * sn;
  x[base + 32 + i] = x2 * c + x1 * sn;
}

// ---------------------------------------------------------------------------
// Causal GQA attention: one wave per (head, query). Online softmax over keys.
// q:[S,16,64]  k,v:[S,4,64]  out:[S,1024]
// ---------------------------------------------------------------------------
__global__ void attn_kernel(const float* __restrict__ q,
                            const float* __restrict__ k,
                            const float* __restrict__ v,
                            const int* __restrict__ mask,
                            float* __restrict__ out) {
  const int gw   = (blockIdx.x * blockDim.x + threadIdx.x) >> 5;
  const int lane = threadIdx.x & 31;
  const int h  = gw / SEQ;
  const int qi = gw % SEQ;
  if (h >= AH) return;
  const int kvh = h >> 2;                     // H/HKV = 4

  const float q0 = q[((size_t)qi * AH + h) * ADH + lane];
  const float q1 = q[((size_t)qi * AH + h) * ADH + 32 + lane];
  const float scale = 0.125f;                 // 1/sqrt(64)

  float m = -3.0e38f, l = 0.0f, acc0 = 0.0f, acc1 = 0.0f;
  for (int kk = 0; kk <= qi; ++kk) {
    const size_t kb = ((size_t)kk * AHKV + kvh) * ADH;
    float p = q0 * k[kb + lane] + q1 * k[kb + 32 + lane];
    #pragma unroll
    for (int off = 16; off > 0; off >>= 1) p += __shfl_xor(p, off, 32);
    float s = p * scale + ((mask[kk] > 0) ? 0.0f : -1.0e9f);
    const float mn = fmaxf(m, s);
    const float corr = __expf(m - mn);
    const float pe   = __expf(s - mn);
    l    = l    * corr + pe;
    acc0 = acc0 * corr + pe * v[kb + lane];
    acc1 = acc1 * corr + pe * v[kb + 32 + lane];
    m = mn;
  }
  const float inv = 1.0f / l;
  out[(size_t)qi * DMODEL + h * ADH + lane]      = acc0 * inv;
  out[(size_t)qi * DMODEL + h * ADH + 32 + lane] = acc1 * inv;
}

// ---------------------------------------------------------------------------
// Mamba depthwise causal conv (width 4) + SiLU over proj[:, 2048:4352]
// ---------------------------------------------------------------------------
__global__ void conv_silu_kernel(const float* __restrict__ proj,
                                 const float* __restrict__ conv_w,  // [2304,4]
                                 const float* __restrict__ conv_b,
                                 float* __restrict__ conv) {
  const int idx = blockIdx.x * blockDim.x + threadIdx.x;
  if (idx >= SEQ * CONVD) return;
  const int c = idx % CONVD;
  const int t = idx / CONVD;
  float acc = conv_b[c];
  #pragma unroll
  for (int kk = 0; kk < 4; ++kk) {
    const int ts = t - 3 + kk;
    if (ts >= 0)
      acc += proj[(size_t)ts * NPROJ + DIN + c] * conv_w[c * 4 + kk];
  }
  conv[(size_t)t * CONVD + c] = silu_f(acc);
}

// ---------------------------------------------------------------------------
// dt = softplus(proj[:,4352+h] + dt_bias);  dA = exp(dt * (-exp(A_log)))
// ---------------------------------------------------------------------------
__global__ void dt_kernel(const float* __restrict__ proj,
                          const float* __restrict__ dt_bias,
                          const float* __restrict__ A_log,
                          float* __restrict__ dt_o, float* __restrict__ dA_o) {
  const int idx = blockIdx.x * blockDim.x + threadIdx.x;
  if (idx >= SEQ * MHEADS) return;
  const int h = idx & 31;
  const int t = idx >> 5;
  const float x = proj[(size_t)t * NPROJ + (2 * DIN + 2 * GN) + h] + dt_bias[h];
  const float sp = (x > 20.f) ? x : __logf(1.f + __expf(x));
  dt_o[idx] = sp;
  dA_o[idx] = __expf(sp * (-__expf(A_log[h])));
}

// ---------------------------------------------------------------------------
// Selective-state scan: block = one head (64 threads), thread = one p row.
// State[p][0..127] lives entirely in VGPRs (128 regs/lane).
// ---------------------------------------------------------------------------
__launch_bounds__(64)
__global__ void mamba_scan_kernel(const float* __restrict__ conv, // [S,2304]
                                  const float* __restrict__ dt,   // [S,32]
                                  const float* __restrict__ dA,   // [S,32]
                                  const float* __restrict__ D_skip,
                                  float* __restrict__ y) {        // [S,2048]
  const int head = blockIdx.x;      // 0..31
  const int p    = threadIdx.x;     // 0..63
  float st[DSTATE];
  #pragma unroll
  for (int n = 0; n < DSTATE; ++n) st[n] = 0.f;
  __shared__ float Bs[DSTATE], Cs[DSTATE];
  const float dsk = D_skip[head];

  for (int t = 0; t < SEQ; ++t) {
    const float* row = conv + (size_t)t * CONVD;
    Bs[p]      = row[DIN + p];
    Bs[p + 64] = row[DIN + p + 64];
    Cs[p]      = row[DIN + GN + p];
    Cs[p + 64] = row[DIN + GN + p + 64];
    __syncthreads();

    const float a  = dA[t * MHEADS + head];
    const float xp = row[head * MDH + p];
    const float c0 = dt[t * MHEADS + head] * xp;
    float acc = 0.f;
    #pragma unroll
    for (int n = 0; n < DSTATE; ++n) {
      st[n] = st[n] * a + c0 * Bs[n];
      acc  += st[n] * Cs[n];
    }
    y[(size_t)t * DIN + head * MDH + p] = acc + xp * dsk;
    __syncthreads();
  }
}

// ---------------------------------------------------------------------------
// out = rmsnorm(y * silu(gate), w) over D_INNER; gate has row stride NPROJ
// ---------------------------------------------------------------------------
__global__ void gated_rmsnorm_kernel(const float* __restrict__ y,
                                     const float* __restrict__ gate,
                                     const float* __restrict__ w,
                                     float* __restrict__ out) {
  const int row = blockIdx.x;
  float ss = 0.f;
  for (int d = threadIdx.x; d < DIN; d += blockDim.x) {
    const float g = gate[(size_t)row * NPROJ + d];
    const float val = y[(size_t)row * DIN + d] * silu_f(g);
    out[(size_t)row * DIN + d] = val;
    ss += val * val;
  }
  __shared__ float red[256];
  red[threadIdx.x] = ss; __syncthreads();
  for (int st = 128; st > 0; st >>= 1) {
    if ((int)threadIdx.x < st) red[threadIdx.x] += red[threadIdx.x + st];
    __syncthreads();
  }
  const float inv = rsqrtf(red[0] / (float)DIN + 1e-6f);
  for (int d = threadIdx.x; d < DIN; d += blockDim.x)
    out[(size_t)row * DIN + d] *= inv * w[d];
}

// ---------------------------------------------------------------------------
// Elementwise helpers
// ---------------------------------------------------------------------------
__global__ void silu_mul_kernel(float* __restrict__ g, const float* __restrict__ u,
                                int n) {
  const int i = blockIdx.x * blockDim.x + threadIdx.x;
  if (i < n) g[i] = silu_f(g[i]) * u[i];
}

__global__ void add_kernel(float* __restrict__ a, const float* __restrict__ b, int n) {
  const int i = blockIdx.x * blockDim.x + threadIdx.x;
  if (i < n) a[i] += b[i];
}

// ---------------------------------------------------------------------------
// Host orchestration
// ---------------------------------------------------------------------------
struct LayerP {
  const float *input_ln_w, *pre_ff_ln_w, *gate_proj, *up_proj, *down_proj;
  const float *in_proj, *conv_w, *conv_b, *dt_bias, *A_log, *D_skip, *mamba_norm_w, *out_proj;
  const float *q_proj, *k_proj, *v_proj, *o_proj;
  int is_mamba;
};

extern "C" void kernel_launch(void* const* d_in, const int* in_sizes, int n_in,
                              void* d_out, int out_size, void* d_ws, size_t ws_size,
                              hipStream_t stream) {
  (void)in_sizes; (void)n_in; (void)out_size; (void)ws_size;

  // ---- unpack inputs in setup_inputs() insertion order ----
  int idx = 0;
  const float* embed        = (const float*)d_in[idx++];
  const float* final_norm_w = (const float*)d_in[idx++];
  const int is_mamba[4] = {1, 1, 0, 1};
  LayerP L[4];
  for (int i = 0; i < 4; ++i) {
    L[i].is_mamba    = is_mamba[i];
    L[i].input_ln_w  = (const float*)d_in[idx++];
    L[i].pre_ff_ln_w = (const float*)d_in[idx++];
    L[i].gate_proj   = (const float*)d_in[idx++];
    L[i].up_proj     = (const float*)d_in[idx++];
    L[i].down_proj   = (const float*)d_in[idx++];
    if (is_mamba[i]) {
      L[i].in_proj      = (const float*)d_in[idx++];
      L[i].conv_w       = (const float*)d_in[idx++];
      L[i].conv_b      = (const float*)d_in[idx++];
      L[i].dt_bias      = (const float*)d_in[idx++];
      L[i].A_log        = (const float*)d_in[idx++];
      L[i].D_skip       = (const float*)d_in[idx++];
      L[i].mamba_norm_w = (const float*)d_in[idx++];
      L[i].out_proj     = (const float*)d_in[idx++];
    } else {
      L[i].q_proj = (const float*)d_in[idx++];
      L[i].k_proj = (const float*)d_in[idx++];
      L[i].v_proj = (const float*)d_in[idx++];
      L[i].o_proj = (const float*)d_in[idx++];
    }
  }
  const int* input_ids = (const int*)d_in[idx++];
  const int* attn_mask = (const int*)d_in[idx++];
  const int* pos_ids   = (const int*)d_in[idx++];

  // ---- carve workspace ----
  float* ws = (float*)d_ws;
  float* h    = ws;                 ws += (size_t)SEQ * DMODEL;
  float* xn   = ws;                 ws += (size_t)SEQ * DMODEL;
  float* proj = ws;                 ws += (size_t)SEQ * NPROJ;
  float* cvb  = ws;                 ws += (size_t)SEQ * CONVD;
  float* yb   = ws;                 ws += (size_t)SEQ * DIN;
  float* zb   = ws;                 ws += (size_t)SEQ * DIN;
  float* gbuf = ws;                 ws += (size_t)SEQ * DFF_;
  float* ubuf = ws;                 ws += (size_t)SEQ * DFF_;
  float* dtb  = ws;                 ws += (size_t)SEQ * MHEADS;
  float* dAb  = ws;                 ws += (size_t)SEQ * MHEADS;
  float* qb   = ws;                 ws += (size_t)SEQ * AH * ADH;
  float* kb   = ws;                 ws += (size_t)SEQ * AHKV * ADH;
  float* vb   = ws;                 ws += (size_t)SEQ * AHKV * ADH;
  float* aob  = ws;                 ws += (size_t)SEQ * DMODEL;
  float* xout = ws;                 ws += (size_t)SEQ * DMODEL;

  const int nD  = SEQ * DMODEL;

  // ---- embedding ----
  embed_kernel<<<SEQ, 256, 0, stream>>>(embed, input_ids, h);

  for (int li = 0; li < 4; ++li) {
    const LayerP& P = L[li];

    // mixer branch
    rmsnorm_kernel<<<SEQ, 256, 0, stream>>>(h, P.input_ln_w, xn, DMODEL);
    if (P.is_mamba) {
      run_gemm<NPROJ, DMODEL>(xn, P.in_proj, proj, stream);
      conv_silu_kernel<<<(SEQ * CONVD + 255) / 256, 256, 0, stream>>>(
          proj, P.conv_w, P.conv_b, cvb);
      dt_kernel<<<(SEQ * MHEADS + 255) / 256, 256, 0, stream>>>(
          proj, P.dt_bias, P.A_log, dtb, dAb);
      mamba_scan_kernel<<<MHEADS, 64, 0, stream>>>(cvb, dtb, dAb, P.D_skip, yb);
      gated_rmsnorm_kernel<<<SEQ, 256, 0, stream>>>(yb, proj, P.mamba_norm_w, zb);
      run_gemm<DMODEL, DIN>(zb, P.out_proj, xout, stream);
    } else {
      run_gemm<AH * ADH, DMODEL>(xn, P.q_proj, qb, stream);
      run_gemm<AHKV * ADH, DMODEL>(xn, P.k_proj, kb, stream);
      run_gemm<AHKV * ADH, DMODEL>(xn, P.v_proj, vb, stream);
      rope_kernel<<<(SEQ * AH * 32 + 255) / 256, 256, 0, stream>>>(qb, pos_ids, AH);
      rope_kernel<<<(SEQ * AHKV * 32 + 255) / 256, 256, 0, stream>>>(kb, pos_ids, AHKV);
      attn_kernel<<<(AH * SEQ) / 8, 256, 0, stream>>>(qb, kb, vb, attn_mask, aob);
      run_gemm<DMODEL, AH * ADH>(aob, P.o_proj, xout, stream);
    }
    add_kernel<<<(nD + 255) / 256, 256, 0, stream>>>(h, xout, nD);

    // MLP branch
    rmsnorm_kernel<<<SEQ, 256, 0, stream>>>(h, P.pre_ff_ln_w, xn, DMODEL);
    run_gemm<DFF_, DMODEL>(xn, P.gate_proj, gbuf, stream);
    run_gemm<DFF_, DMODEL>(xn, P.up_proj,   ubuf, stream);
    silu_mul_kernel<<<(SEQ * DFF_ + 255) / 256, 256, 0, stream>>>(gbuf, ubuf, SEQ * DFF_);
    run_gemm<DMODEL, DFF_>(gbuf, P.down_proj, xout, stream);
    add_kernel<<<(nD + 255) / 256, 256, 0, stream>>>(h, xout, nD);
  }

  // ---- final norm -> output (fp32) ----
  rmsnorm_kernel<<<SEQ, 256, 0, stream>>>(h, final_norm_w, (float*)d_out, DMODEL);
}
